// Feast_GCN_37744172597477
// MI455X (gfx1250) — matrix operand order repaired and checked by reading, not compiled, not run on hardware
//
#include <hip/hip_runtime.h>
#include <hip/hip_bf16.h>

#define HEADS 6

typedef __attribute__((ext_vector_type(16))) __bf16 v16bf;
typedef __attribute__((ext_vector_type(8)))  __bf16 v8bf;
typedef __attribute__((ext_vector_type(4)))  __bf16 v4bf;
typedef __attribute__((ext_vector_type(8)))  float  v8f;

static inline int cdiv(int a, int b) { return (a + b - 1) / b; }
static inline size_t alignup(size_t x) { return (x + 255) & ~(size_t)255; }

__device__ __forceinline__ int dmin(int a, int b) { return a < b ? a : b; }
// low 32 bits of a generic pointer into LDS == LDS byte offset (ISA 10.2: addr[31:0])
__device__ __forceinline__ unsigned lds_off(const void* p) {
  return (unsigned)(unsigned long long)p;
}

#define ASYNC16(lds, gp)                                                     \
  asm volatile("global_load_async_to_lds_b128 %0, %1, off" ::"v"(lds),       \
               "v"(gp)                                                       \
               : "memory")

// -------------------------------------------------------------------------
// utility kernels
// -------------------------------------------------------------------------
__global__ void k_zero(float* __restrict__ p, int n) {
  int i = blockIdx.x * blockDim.x + threadIdx.x;
  if (i < n) p[i] = 0.0f;
}

// x0[n][0..31] bf16, = [pos||norm] zero-padded to stride 32 (so GEMM K=32 path works)
__global__ void k_concat(const float* __restrict__ pos, const float* __restrict__ nrm,
                         __bf16* __restrict__ x0, int N) {
  int i = blockIdx.x * blockDim.x + threadIdx.x;
  if (i < N) {
    __bf16* r = x0 + (size_t)i * 32;
    r[0] = (__bf16)pos[i * 3 + 0];
    r[1] = (__bf16)pos[i * 3 + 1];
    r[2] = (__bf16)pos[i * 3 + 2];
    r[3] = (__bf16)nrm[i * 3 + 0];
    r[4] = (__bf16)nrm[i * 3 + 1];
    r[5] = (__bf16)nrm[i * 3 + 2];
#pragma unroll
    for (int k = 6; k < 32; ++k) r[k] = (__bf16)0.0f;
  }
}

// Wt[col][k] (bf16, [NCp][Kp], zero-padded in both dims) = W[k][col] (f32, [K][Ncols])
__global__ void k_prepW(const float* __restrict__ W, __bf16* __restrict__ Wt,
                        int K, int Kp, int Ncols, int NCp) {
  int idx = blockIdx.x * blockDim.x + threadIdx.x;
  if (idx < NCp * Kp) {
    int col = idx / Kp, k = idx % Kp;
    float v = (k < K && col < Ncols) ? W[(size_t)k * Ncols + col] : 0.0f;
    Wt[idx] = (__bf16)v;
  }
}

// in-degree of each dst node (edge_index row 1)
__global__ void k_count(const long long* __restrict__ ei, float* __restrict__ cnt, int E) {
  int i = blockIdx.x * blockDim.x + threadIdx.x;
  if (i < E) atomicAdd(&cnt[(int)ei[(size_t)E + i]], 1.0f);
}

// t[n,0..5] = x[n,:] @ u   (x bf16, row stride lda; one thread per node)
__global__ void k_tproj(const __bf16* __restrict__ x, int lda, const float* __restrict__ u,
                        float* __restrict__ t, int N, int K) {
  int n = blockIdx.x * blockDim.x + threadIdx.x;
  if (n >= N) return;
  float s[HEADS] = {0.f, 0.f, 0.f, 0.f, 0.f, 0.f};
  const __bf16* row = x + (size_t)n * lda;
  const int K4 = K & ~3;
  for (int k = 0; k < K4; k += 4) {
    v4bf xv = *(const v4bf*)(row + k);
#pragma unroll
    for (int j = 0; j < 4; ++j) {
      float xf = (float)xv[j];
#pragma unroll
      for (int h = 0; h < HEADS; ++h) s[h] += xf * u[(k + j) * HEADS + h];
    }
  }
  for (int k = K4; k < K; ++k) {
    float xf = (float)row[k];
#pragma unroll
    for (int h = 0; h < HEADS; ++h) s[h] += xf * u[k * HEADS + h];
  }
#pragma unroll
  for (int h = 0; h < HEADS; ++h) t[n * HEADS + h] = s[h];
}

// -------------------------------------------------------------------------
// Tiled WMMA GEMM:  Y[Mp,NCOLS](bf16) = A[M,K](bf16) @ Bt[NCOLS,K](bf16)^T
// NCOLS: compile-time, multiple of 128.  K: runtime, multiple of 32.
// Y has Mp = ceil(M/128)*128 padded rows -> epilogue is branch-free.
// Tiles staged via double-buffered global_load_async_to_lds_b128; ASYNCcnt
// completes in order, so waiting <=4 after issuing the next tile guarantees
// the current tile is resident.
// -------------------------------------------------------------------------
template <int NCOLS>
__global__ __launch_bounds__(256) void k_gemm_bf16(const __bf16* __restrict__ A,
                                                   const __bf16* __restrict__ Bt,
                                                   __bf16* __restrict__ Y,
                                                   int M, int K) {
  __shared__ __align__(16) __bf16 As[2][128 * 40];  // [row][k], row stride 40 elems (80B)
  __shared__ __align__(16) __bf16 Bs[2][128 * 40];  // [col][k] (B transposed)

  const int tid  = threadIdx.x;
  const int lane = tid & 31;
  const int wave = tid >> 5;
  const int wm   = wave & 1;     // 64-row block within tile
  const int wn   = wave >> 1;    // 32-col block within tile
  const int hi   = lane >> 4;    // half-wave select
  const int l16  = lane & 15;
  const int m0   = blockIdx.x * 128;
  const int n0   = blockIdx.y * 128;

  // staging: 512 chunks of 16B per tile; chunk c -> row c>>2, part c&3
  const int c0 = 2 * tid, c1 = 2 * tid + 1;
  const int r0 = c0 >> 2, p0 = c0 & 3;
  const int r1 = c1 >> 2, p1 = c1 & 3;
  const __bf16* gA0 = A + (size_t)dmin(m0 + r0, M - 1) * K + p0 * 8;  // row clamp
  const __bf16* gA1 = A + (size_t)dmin(m0 + r1, M - 1) * K + p1 * 8;
  const __bf16* gB0 = Bt + (size_t)(n0 + r0) * K + p0 * 8;            // NCOLS padded
  const __bf16* gB1 = Bt + (size_t)(n0 + r1) * K + p1 * 8;
  const unsigned lA0 = lds_off(As) + r0 * 80 + p0 * 16;
  const unsigned lA1 = lds_off(As) + r1 * 80 + p1 * 16;
  const unsigned lB0 = lds_off(Bs) + r0 * 80 + p0 * 16;
  const unsigned lB1 = lds_off(Bs) + r1 * 80 + p1 * 16;
  const unsigned BUFB = 128 * 40 * 2;  // bytes per LDS buffer

  v8f acc[4][2];
#pragma unroll
  for (int ms = 0; ms < 4; ++ms)
#pragma unroll
    for (int ns = 0; ns < 2; ++ns)
      acc[ms][ns] = (v8f){0.f, 0.f, 0.f, 0.f, 0.f, 0.f, 0.f, 0.f};

  const int niter = K >> 5;
  // prologue: tile 0 -> buffer 0
  ASYNC16(lA0, gA0);
  ASYNC16(lA1, gA1);
  ASYNC16(lB0, gB0);
  ASYNC16(lB1, gB1);

  for (int it = 0; it < niter; ++it) {
    const int cur = it & 1;
    if (it + 1 < niter) {  // issue next tile into the other buffer
      const unsigned bo = (unsigned)((it + 1) & 1) * BUFB;
      const int ko = (it + 1) * 32;
      ASYNC16(lA0 + bo, gA0 + ko);
      ASYNC16(lA1 + bo, gA1 + ko);
      ASYNC16(lB0 + bo, gB0 + ko);
      ASYNC16(lB1 + bo, gB1 + ko);
      asm volatile("s_wait_asynccnt 0x4" ::: "memory");
    } else {
      asm volatile("s_wait_asynccnt 0x0" ::: "memory");
    }
    __syncthreads();

    // A fragment (16x32 bf16): lane half 'hi' holds K = {0..7,16..23} or {8..15,24..31}
    v16bf a[4];
#pragma unroll
    for (int ms = 0; ms < 4; ++ms) {
      const __bf16* base = &As[cur][(wm * 64 + ms * 16 + l16) * 40];
      ((v8bf*)&a[ms])[0] = *(const v8bf*)(base + hi * 8);
      ((v8bf*)&a[ms])[1] = *(const v8bf*)(base + 16 + hi * 8);
    }
    // B fragment (32x16 bf16): lane half 'hi' holds K = {0..7,16..23} or {8..15,24..31}, N = l16
    v16bf b[2];
#pragma unroll
    for (int ns = 0; ns < 2; ++ns) {
      const __bf16* base = &Bs[cur][(wn * 32 + ns * 16 + l16) * 40];
      ((v8bf*)&b[ns])[0] = *(const v8bf*)(base + hi * 8);
      ((v8bf*)&b[ns])[1] = *(const v8bf*)(base + 16 + hi * 8);
    }

#pragma unroll
    for (int ms = 0; ms < 4; ++ms)
#pragma unroll
      for (int ns = 0; ns < 2; ++ns)
        acc[ms][ns] = __builtin_amdgcn_wmma_f32_16x16x32_bf16(
            false, a[ms], false, b[ns], (short)0, acc[ms][ns], false, false);

    __syncthreads();
  }

  // branch-free epilogue: C/D layout => VGPR r : M = r + 8*hi, N = l16
#pragma unroll
  for (int ms = 0; ms < 4; ++ms) {
#pragma unroll
    for (int ns = 0; ns < 2; ++ns) {
      const int gn = n0 + wn * 32 + ns * 16 + l16;
      __bf16* yp = Y + (size_t)(m0 + wm * 64 + ms * 16 + hi * 8) * NCOLS + gn;
#pragma unroll
      for (int r = 0; r < 8; ++r) yp[(size_t)r * NCOLS] = (__bf16)acc[ms][ns][r];
    }
  }
}

// -------------------------------------------------------------------------
// Edge kernel, out=128: one wave32 per edge, lane owns channels lane*4..lane*4+3
// q = softmax_h(t[src]-t[dst]+c);  m = sum_h q_h * Y[src, h*128 + ch];  atomic scatter.
// Each head is one 8B (v4bf) load; wave reads 256B contiguous per head.
// -------------------------------------------------------------------------
__global__ __launch_bounds__(256) void k_edge128(const long long* __restrict__ ei,
                                                 const float* __restrict__ t,
                                                 const float* __restrict__ cvec,
                                                 const __bf16* __restrict__ Y,
                                                 float* __restrict__ acc, int E) {
  int gid  = blockIdx.x * blockDim.x + threadIdx.x;
  int edge = gid >> 5;            // wave-uniform (wave32)
  int lane = gid & 31;
  if (edge >= E) return;

  int src = (int)ei[edge];
  int dst = (int)ei[(size_t)E + edge];

  float lg[HEADS];
  float mx = -1e30f;
#pragma unroll
  for (int h = 0; h < HEADS; ++h) {
    lg[h] = t[src * HEADS + h] - t[dst * HEADS + h] + cvec[h];
    mx = fmaxf(mx, lg[h]);
  }
  float s = 0.0f;
#pragma unroll
  for (int h = 0; h < HEADS; ++h) { lg[h] = __expf(lg[h] - mx); s += lg[h]; }
  const float inv = 1.0f / s;

  const __bf16* y = Y + (size_t)src * (HEADS * 128) + lane * 4;
  float m0 = 0.f, m1 = 0.f, m2 = 0.f, m3 = 0.f;
#pragma unroll
  for (int h = 0; h < HEADS; ++h) {
    float q = lg[h] * inv;
    v4bf yv = *(const v4bf*)(y + h * 128);
    m0 += q * (float)yv[0];
    m1 += q * (float)yv[1];
    m2 += q * (float)yv[2];
    m3 += q * (float)yv[3];
  }
  float* a = acc + (size_t)dst * 128 + lane * 4;
  atomicAdd(a + 0, m0);
  atomicAdd(a + 1, m1);
  atomicAdd(a + 2, m2);
  atomicAdd(a + 3, m3);
}

// Edge kernel, out=3 (last layer): one thread per edge.  Y4 row stride = 128
// (layer-4 GEMM writes the 18 real columns into a 128-wide padded matrix).
__global__ void k_edge3(const long long* __restrict__ ei, const float* __restrict__ t,
                        const float* __restrict__ cvec, const __bf16* __restrict__ Y,
                        float* __restrict__ acc, int E) {
  int e = blockIdx.x * blockDim.x + threadIdx.x;
  if (e >= E) return;
  int src = (int)ei[e];
  int dst = (int)ei[(size_t)E + e];

  float lg[HEADS];
  float mx = -1e30f;
#pragma unroll
  for (int h = 0; h < HEADS; ++h) {
    lg[h] = t[src * HEADS + h] - t[dst * HEADS + h] + cvec[h];
    mx = fmaxf(mx, lg[h]);
  }
  float s = 0.0f;
#pragma unroll
  for (int h = 0; h < HEADS; ++h) { lg[h] = __expf(lg[h] - mx); s += lg[h]; }
  const float inv = 1.0f / s;

  const __bf16* y = Y + (size_t)src * 128;
#pragma unroll
  for (int c = 0; c < 3; ++c) {
    float m = 0.0f;
#pragma unroll
    for (int h = 0; h < HEADS; ++h) m += (lg[h] * inv) * (float)y[h * 3 + c];
    atomicAdd(&acc[(size_t)dst * 3 + c], m);
  }
}

// mean + bias + ReLU (128-channel layers), bf16 activation output
__global__ void k_finalize128(const float* __restrict__ acc, const float* __restrict__ cnt,
                              const float* __restrict__ b, __bf16* __restrict__ xout, int N) {
  int i = blockIdx.x * blockDim.x + threadIdx.x;
  if (i < N * 128) {
    int n = i >> 7, ch = i & 127;
    float d = cnt[n];
    d = d < 1.0f ? 1.0f : d;
    float v = acc[i] / d + b[ch];
    xout[i] = (__bf16)(v > 0.0f ? v : 0.0f);
  }
}

// mean + bias, no ReLU (final 3-channel layer, f32 output)
__global__ void k_finalize3(const float* __restrict__ acc, const float* __restrict__ cnt,
                            const float* __restrict__ b, float* __restrict__ out, int N) {
  int i = blockIdx.x * blockDim.x + threadIdx.x;
  if (i < N * 3) {
    int n = i / 3, ch = i % 3;
    float d = cnt[n];
    d = d < 1.0f ? 1.0f : d;
    out[i] = acc[i] / d + b[ch];
  }
}

// -------------------------------------------------------------------------
// host-side layer driver (128-out layers).  Kp = padded K (multiple of 32).
// -------------------------------------------------------------------------
static void run_layer128(const __bf16* xin, int K, int Kp, const float* W, const float* u,
                         const float* cvec, const float* b, const long long* ei,
                         float* tbuf, __bf16* Wt, __bf16* Ybf, float* acc,
                         const float* cnt, __bf16* xout, int N, int E, hipStream_t s) {
  const int NC = HEADS * 128;  // 768, multiple of 128
  k_tproj<<<cdiv(N, 256), 256, 0, s>>>(xin, Kp, u, tbuf, N, K);
  k_prepW<<<cdiv(NC * Kp, 256), 256, 0, s>>>(W, Wt, K, Kp, NC, NC);
  dim3 gg(cdiv(N, 128), NC / 128);
  k_gemm_bf16<768><<<gg, 256, 0, s>>>(xin, Wt, Ybf, N, Kp);
  k_zero<<<cdiv(N * 128, 256), 256, 0, s>>>(acc, N * 128);
  k_edge128<<<cdiv(E * 32, 256), 256, 0, s>>>(ei, tbuf, cvec, Ybf, acc, E);
  k_finalize128<<<cdiv(N * 128, 256), 256, 0, s>>>(acc, cnt, b, xout, N);
}

extern "C" void kernel_launch(void* const* d_in, const int* in_sizes, int n_in,
                              void* d_out, int out_size, void* d_ws, size_t ws_size,
                              hipStream_t stream) {
  const int N  = in_sizes[0] / 3;
  const int E  = in_sizes[2] / 2;
  const int Mp = cdiv(N, 128) * 128;   // padded rows for GEMM outputs

  const float* pos = (const float*)d_in[0];
  const float* nrm = (const float*)d_in[1];
  const long long* ei = (const long long*)d_in[2];   // int64 in reference
  const float* w1 = (const float*)d_in[3];
  const float* u1 = (const float*)d_in[4];
  const float* c1 = (const float*)d_in[5];
  const float* b1 = (const float*)d_in[6];
  const float* w2 = (const float*)d_in[7];
  const float* u2 = (const float*)d_in[8];
  const float* c2 = (const float*)d_in[9];
  const float* b2 = (const float*)d_in[10];
  const float* w3 = (const float*)d_in[11];
  const float* u3 = (const float*)d_in[12];
  const float* c3 = (const float*)d_in[13];
  const float* b3 = (const float*)d_in[14];
  const float* w4 = (const float*)d_in[15];
  const float* u4 = (const float*)d_in[16];
  const float* c4 = (const float*)d_in[17];
  const float* b4 = (const float*)d_in[18];

  // workspace layout
  char* ws = (char*)d_ws;
  size_t off = 0;
  __bf16* x0   = (__bf16*)(ws + off); off += alignup((size_t)N * 32 * 2);
  __bf16* xA   = (__bf16*)(ws + off); off += alignup((size_t)N * 128 * 2);
  __bf16* xB   = (__bf16*)(ws + off); off += alignup((size_t)N * 128 * 2);
  __bf16* Ybf  = (__bf16*)(ws + off); off += alignup((size_t)Mp * HEADS * 128 * 2);
  __bf16* Wt   = (__bf16*)(ws + off); off += alignup((size_t)HEADS * 128 * 128 * 2);
  float*  tbuf = (float*)(ws + off);  off += alignup((size_t)N * HEADS * 4);
  float*  acc  = (float*)(ws + off);  off += alignup((size_t)N * 128 * 4);
  float*  cnt  = (float*)(ws + off);  off += alignup((size_t)N * 4);
  (void)ws_size; (void)n_in; (void)out_size;

  // x0 = [pos || norm] (bf16, zero-padded to stride 32)
  k_concat<<<cdiv(N, 256), 256, 0, stream>>>(pos, nrm, x0, N);

  // in-degree (same for all layers)
  k_zero<<<cdiv(N, 256), 256, 0, stream>>>(cnt, N);
  k_count<<<cdiv(E, 256), 256, 0, stream>>>(ei, cnt, E);

  // layers 1..3 (ReLU, out=128)
  run_layer128(x0, 6,   32,  w1, u1, c1, b1, ei, tbuf, Wt, Ybf, acc, cnt, xA, N, E, stream);
  run_layer128(xA, 128, 128, w2, u2, c2, b2, ei, tbuf, Wt, Ybf, acc, cnt, xB, N, E, stream);
  run_layer128(xB, 128, 128, w3, u3, c3, b3, ei, tbuf, Wt, Ybf, acc, cnt, xA, N, E, stream);

  // layer 4 (out=3, no ReLU) -> d_out.  18 real cols padded to 128.
  float* outp = (float*)d_out;
  k_tproj<<<cdiv(N, 256), 256, 0, stream>>>(xA, 128, u4, tbuf, N, 128);
  k_prepW<<<cdiv(128 * 128, 256), 256, 0, stream>>>(w4, Wt, 128, 128, HEADS * 3, 128);
  dim3 gg4(cdiv(N, 128), 1);
  k_gemm_bf16<128><<<gg4, 256, 0, stream>>>(xA, Wt, Ybf, N, 128);
  k_zero<<<cdiv(N * 3, 256), 256, 0, stream>>>(acc, N * 3);
  k_edge3<<<cdiv(E, 256), 256, 0, stream>>>(ei, tbuf, c4, Ybf, acc, E);
  k_finalize3<<<cdiv(N * 3, 256), 256, 0, stream>>>(acc, cnt, b4, outp, N);
}
